// Mamba2Block_27882927685715
// MI455X (gfx1250) — matrix operand order repaired and checked by reading, not compiled, not run on hardware
//
#include <hip/hip_runtime.h>
#include <hip/hip_bf16.h>

// ---------------------------------------------------------------------------
// Mamba2 block forward for MI455X (gfx1250, wave32, WMMA).
// All matrix-shaped work uses v_wmma_f32_16x16x32_bf16 (bf16 operands, f32 acc).
// Big GEMMs: 128x64 block tiles, double-buffered LDS, global prefetch.
// ---------------------------------------------------------------------------

typedef __bf16 bf16;
typedef __attribute__((ext_vector_type(16))) __bf16 v16bf;
typedef __attribute__((ext_vector_type(8)))  float  v8f;

#define D_MODEL 2048
#define D_INNER 4096
#define NH      64
#define D_HEAD  64
#define SN      128      // state dim N
#define NG      8        // groups
#define CHUNK   256
#define GN      1024     // NG*SN
#define CONV_DIM 6144
#define IN_PROJ 10304
#define LSEQ    2048
#define BB      2
#define NCH     8        // chunks per sequence
#define ROWS    (BB*LSEQ)    // 4096 tokens
#define BCH     (BB*NCH*NH)  // 1024 (b,c,h) triples

__device__ inline v8f wmma_bf16(v16bf a, v16bf b, v8f c) {
  return __builtin_amdgcn_wmma_f32_16x16x32_bf16(false, a, false, b, (short)0, c,
                                                 false, false);
}

// A-operand fragment (16x32, 16-bit): lane l<16 holds row m=l with K {0..15},
// lane l>=16 holds row m=l-16 with K {16..31}.  smem[m*ld + k].
__device__ inline v16bf frag_a(const bf16* s, int ld, int lane) {
  const int half = lane >> 4, m = lane & 15;
  v16bf a;
#pragma unroll
  for (int i = 0; i < 16; ++i) a[i] = s[m * ld + half * 16 + i];
  return a;
}

// B-operand (32x16) from LDS staged as [k][n] rows: b[i] = B[half*16+i][n]
__device__ inline v16bf frag_b_kn(const bf16* s, int ld, int lane) {
  const int half = lane >> 4, n = lane & 15;
  v16bf b;
#pragma unroll
  for (int i = 0; i < 16; ++i) b[i] = s[(half * 16 + i) * ld + n];
  return b;
}

// B-operand (32x16) from LDS staged as [n][k] rows (NT): b[i] = s[n*ld + half*16+i]
__device__ inline v16bf frag_b_nk(const bf16* s, int ld, int lane) {
  const int half = lane >> 4, n = lane & 15;
  v16bf b;
#pragma unroll
  for (int i = 0; i < 16; ++i) b[i] = s[n * ld + half * 16 + i];
  return b;
}

#define V8F_ZERO {0.f,0.f,0.f,0.f,0.f,0.f,0.f,0.f}

// ---------------------------------------------------------------------------
// 1. RMSNorm(x) -> bf16
// ---------------------------------------------------------------------------
__global__ __launch_bounds__(256)
void rmsnorm_x_kernel(const float* __restrict__ x, const float* __restrict__ w,
                      bf16* __restrict__ out) {
  __shared__ float red[256];
  const int row = blockIdx.x;
  const float* xr = x + (size_t)row * D_MODEL;
  float ss = 0.f;
  for (int i = threadIdx.x; i < D_MODEL; i += 256) { float v = xr[i]; ss += v * v; }
  red[threadIdx.x] = ss; __syncthreads();
  for (int s = 128; s > 0; s >>= 1) {
    if (threadIdx.x < s) red[threadIdx.x] += red[threadIdx.x + s];
    __syncthreads();
  }
  const float rms = rsqrtf(red[0] / (float)D_MODEL + 1e-6f);
  for (int i = threadIdx.x; i < D_MODEL; i += 256)
    out[(size_t)row * D_MODEL + i] = (bf16)(xr[i] * rms * w[i]);
}

__global__ void f32_to_bf16_kernel(const float* __restrict__ in,
                                   bf16* __restrict__ out, size_t n) {
  size_t i = (size_t)blockIdx.x * blockDim.x + threadIdx.x;
  if (i < n) out[i] = (bf16)in[i];
}

// ---------------------------------------------------------------------------
// 2. Generic NT GEMM: C[m,n] = sum_k A[m,k]*B[n,k] (+ Cadd), A/B bf16, C f32.
//    128x64 block tile, BK=32, double-buffered LDS, 256 threads = 8 waves,
//    4 accumulators / wave, one barrier per K-step, prefetch 2 steps ahead.
// ---------------------------------------------------------------------------
__global__ __launch_bounds__(256)
void gemm_bf16_nt(const bf16* __restrict__ A, const bf16* __restrict__ B,
                  const float* __restrict__ Cadd, float* __restrict__ C,
                  int K, int lda, int ldb, int ldc) {
  __shared__ bf16 As[2][128 * 40];
  __shared__ bf16 Bs[2][64 * 40];
  const int tid = threadIdx.x, lane = tid & 31, wave = tid >> 5;
  const int m0 = blockIdx.x * 128, n0 = blockIdx.y * 64;
  const int sn = (wave & 3) * 16;          // n-subtile (0..48)
  const int sm = (wave >> 2) * 64;         // 4 m-subtiles: sm, sm+16, sm+32, sm+48
  const int ar = tid >> 1, ac = (tid & 1) * 16;   // A stage: 128 rows x 32 k
  const int br = tid >> 2, bc = (tid & 3) * 8;    // B stage: 64 rows x 32 k

  v8f acc[4] = {V8F_ZERO, V8F_ZERO, V8F_ZERO, V8F_ZERO};

  auto load_stage = [&](int buf, int k0) {
    const bf16* ga = A + (size_t)(m0 + ar) * lda + k0 + ac;
#pragma unroll
    for (int i = 0; i < 16; ++i) As[buf][ar * 40 + ac + i] = ga[i];
    const bf16* gb = B + (size_t)(n0 + br) * ldb + k0 + bc;
#pragma unroll
    for (int i = 0; i < 8; ++i) Bs[buf][br * 40 + bc + i] = gb[i];
  };

  load_stage(0, 0);
  __syncthreads();
  int buf = 0;
  for (int k0 = 0; k0 < K; k0 += 32) {
    if (k0 + 32 < K) {
      load_stage(buf ^ 1, k0 + 32);
      if (k0 + 64 < K) {
        __builtin_prefetch(A + (size_t)(m0 + ar) * lda + k0 + 64 + ac, 0, 1);
        __builtin_prefetch(B + (size_t)(n0 + br) * ldb + k0 + 64 + bc, 0, 1);
      }
    }
    v16bf bfrag = frag_b_nk(&Bs[buf][sn * 40], 40, lane);
#pragma unroll
    for (int i = 0; i < 4; ++i)
      acc[i] = wmma_bf16(frag_a(&As[buf][(sm + i * 16) * 40], 40, lane), bfrag, acc[i]);
    __syncthreads();
    buf ^= 1;
  }
  const int half = lane >> 4, nn = lane & 15;
  const int col = n0 + sn + nn;
#pragma unroll
  for (int i = 0; i < 4; ++i) {
#pragma unroll
    for (int r = 0; r < 8; ++r) {
      const size_t idx = (size_t)(m0 + sm + i * 16 + half * 8 + r) * ldc + col;
      C[idx] = acc[i][r] + (Cadd ? Cadd[idx] : 0.f);
    }
  }
}

// ---------------------------------------------------------------------------
// 3. Fused causal depthwise conv(4) + RMSNorm(conv_norm_w) + SiLU -> bf16
// ---------------------------------------------------------------------------
__global__ __launch_bounds__(256)
void conv_norm_silu_kernel(const float* __restrict__ proj,
                           const float* __restrict__ cw,
                           const float* __restrict__ cb,
                           const float* __restrict__ nw,
                           bf16* __restrict__ out) {
  __shared__ float red[256];
  const int bl = blockIdx.x;               // b*2048 + l
  const int b = bl >> 11, l = bl & 2047;
  float vals[24];
  float ss = 0.f;
#pragma unroll
  for (int ii = 0; ii < 24; ++ii) {
    const int c = threadIdx.x + ii * 256;            // 0..6143
    const int src = (c < D_INNER) ? c : c + NH;      // skip A_dt slice
    float acc = cb[c];
#pragma unroll
    for (int j = 0; j < 4; ++j) {
      const int lp = l - 3 + j;
      if (lp >= 0)
        acc += cw[c * 4 + j] * proj[(size_t)(b * LSEQ + lp) * IN_PROJ + src];
    }
    vals[ii] = acc;
    ss += acc * acc;
  }
  red[threadIdx.x] = ss; __syncthreads();
  for (int s = 128; s > 0; s >>= 1) {
    if (threadIdx.x < s) red[threadIdx.x] += red[threadIdx.x + s];
    __syncthreads();
  }
  const float rms = rsqrtf(red[0] / (float)CONV_DIM + 1e-6f);
#pragma unroll
  for (int ii = 0; ii < 24; ++ii) {
    const int c = threadIdx.x + ii * 256;
    float v = vals[ii] * rms * nw[c];
    v = v / (1.f + __expf(-v));
    out[(size_t)bl * CONV_DIM + c] = (bf16)v;
  }
}

// ---------------------------------------------------------------------------
// 4. dt = softplus(A_dt @ dt_w^T + dt_b);  A = -exp(A_log)*dt
// ---------------------------------------------------------------------------
__global__ __launch_bounds__(64)
void dt_a_kernel(const float* __restrict__ proj, const float* __restrict__ dtw,
                 const float* __restrict__ dtb, const float* __restrict__ alog,
                 float* __restrict__ Aout) {
  const int row = blockIdx.x, h = threadIdx.x;
  const float* ad = proj + (size_t)row * IN_PROJ + D_INNER;
  float s = dtb[h];
  for (int j = 0; j < NH; ++j) s += ad[j] * dtw[h * NH + j];
  const float sp = (s > 20.f) ? s : log1pf(__expf(s));
  Aout[(size_t)row * NH + h] = -__expf(alog[h]) * sp;
}

// ---------------------------------------------------------------------------
// 5. Per-(b,c,h) inclusive cumsum of A over the chunk -> Acs[bch][t]
// ---------------------------------------------------------------------------
__global__ __launch_bounds__(256)
void cumsum_kernel(const float* __restrict__ Aval, float* __restrict__ Acs) {
  __shared__ float buf[256];
  const int bch = blockIdx.x;
  const int h = bch & 63, bc = bch >> 6;   // bc = b*8 + c
  const int t = threadIdx.x;
  buf[t] = Aval[(size_t)(bc * CHUNK + t) * NH + h];
  __syncthreads();
  for (int off = 1; off < 256; off <<= 1) {
    const float add = (t >= off) ? buf[t - off] : 0.f;
    __syncthreads();
    buf[t] += add;
    __syncthreads();
  }
  Acs[(size_t)bch * CHUNK + t] = buf[t];
}

// ---------------------------------------------------------------------------
// 6. M[t,s] = tril * exp(Acs[t]-Acs[s]) * (C[t,:]·B[s,:])  -> bf16
//    Per (bch, 64x64 tile of 256x256), WMMA over K=128.
// ---------------------------------------------------------------------------
__global__ __launch_bounds__(256)
void cb_mask_kernel(const bf16* __restrict__ conv, const float* __restrict__ Acs,
                    bf16* __restrict__ Mmask) {
  __shared__ bf16 Cs[64 * 136];
  __shared__ bf16 Bs[64 * 136];
  __shared__ float at[64], asv[64];
  const int bch = blockIdx.x, tile = blockIdx.y;
  const int tt = (tile >> 2) * 64, ts = (tile & 3) * 64;
  const int h = bch & 63, bc = bch >> 6, g = h >> 3;
  const int rowbase = bc * CHUNK;
  {
    const int r = threadIdx.x >> 2;
    const int c0 = (threadIdx.x & 3) * 32;
    const bf16* gc = conv + (size_t)(rowbase + tt + r) * CONV_DIM + (D_INNER + GN) + g * SN + c0;
    const bf16* gb = conv + (size_t)(rowbase + ts + r) * CONV_DIM + D_INNER + g * SN + c0;
#pragma unroll
    for (int i = 0; i < 32; ++i) { Cs[r * 136 + c0 + i] = gc[i]; Bs[r * 136 + c0 + i] = gb[i]; }
  }
  if (threadIdx.x < 64)       at[threadIdx.x]       = Acs[(size_t)bch * CHUNK + tt + threadIdx.x];
  else if (threadIdx.x < 128) asv[threadIdx.x - 64] = Acs[(size_t)bch * CHUNK + ts + (threadIdx.x - 64)];
  __syncthreads();

  const int lane = threadIdx.x & 31, wave = threadIdx.x >> 5;
  const int sn = (wave & 3) * 16, sm0 = (wave >> 2) * 32;
  v8f acc0 = V8F_ZERO, acc1 = V8F_ZERO;
#pragma unroll
  for (int k0 = 0; k0 < SN; k0 += 32) {
    v16bf bf = frag_b_nk(Bs + sn * 136 + k0, 136, lane);
    acc0 = wmma_bf16(frag_a(Cs + sm0 * 136 + k0, 136, lane), bf, acc0);
    acc1 = wmma_bf16(frag_a(Cs + (sm0 + 16) * 136 + k0, 136, lane), bf, acc1);
  }
  const int half = lane >> 4, nn = lane & 15;
  const int sloc = ts + sn + nn;
  const float a_s = asv[sn + nn];
#pragma unroll
  for (int r = 0; r < 8; ++r) {
    const int la = sm0 + half * 8 + r, lb = la + 16;
    const int t0 = tt + la, t1 = tt + lb;
    const float v0 = (t0 >= sloc) ? __expf(at[la] - a_s) * acc0[r] : 0.f;
    const float v1 = (t1 >= sloc) ? __expf(at[lb] - a_s) * acc1[r] : 0.f;
    Mmask[(size_t)bch * (CHUNK * CHUNK) + (size_t)t0 * CHUNK + sloc] = (bf16)v0;
    Mmask[(size_t)bch * (CHUNK * CHUNK) + (size_t)t1 * CHUNK + sloc] = (bf16)v1;
  }
}

// ---------------------------------------------------------------------------
// 7. Y_diag[t,d] = sum_s M[t,s]*X[s,d]   (K = 256)
// ---------------------------------------------------------------------------
__global__ __launch_bounds__(256)
void ydiag_kernel(const bf16* __restrict__ Mmask, const bf16* __restrict__ conv,
                  float* __restrict__ Y) {
  __shared__ bf16 As[64 * 40];
  __shared__ bf16 Xs[32 * 72];
  const int bch = blockIdx.x, mt = blockIdx.y;
  const int h = bch & 63, bc = bch >> 6;
  const int t0 = mt * 64;
  const int lane = threadIdx.x & 31, wave = threadIdx.x >> 5;
  const int sn = (wave & 3) * 16, sm0 = (wave >> 2) * 32;
  v8f acc0 = V8F_ZERO, acc1 = V8F_ZERO;
  for (int k0 = 0; k0 < CHUNK; k0 += 32) {
    {
      const int r = threadIdx.x >> 2, c = (threadIdx.x & 3) * 8;
      const bf16* g = Mmask + (size_t)bch * (CHUNK * CHUNK) + (size_t)(t0 + r) * CHUNK + k0 + c;
#pragma unroll
      for (int i = 0; i < 8; ++i) As[r * 40 + c + i] = g[i];
    }
    {
      const int r = threadIdx.x >> 3, c = (threadIdx.x & 7) * 8;
      const bf16* g = conv + (size_t)(bc * CHUNK + k0 + r) * CONV_DIM + h * D_HEAD + c;
#pragma unroll
      for (int i = 0; i < 8; ++i) Xs[r * 72 + c + i] = g[i];
    }
    __syncthreads();
    v16bf bf = frag_b_kn(Xs + sn, 72, lane);
    acc0 = wmma_bf16(frag_a(As + sm0 * 40, 40, lane), bf, acc0);
    acc1 = wmma_bf16(frag_a(As + (sm0 + 16) * 40, 40, lane), bf, acc1);
    __syncthreads();
  }
  const int half = lane >> 4, nn = lane & 15, d = sn + nn;
#pragma unroll
  for (int r = 0; r < 8; ++r) {
    const int ta = t0 + sm0 + half * 8 + r, tb = ta + 16;
    Y[(size_t)(bc * CHUNK + ta) * D_INNER + h * D_HEAD + d] = acc0[r];
    Y[(size_t)(bc * CHUNK + tb) * D_INNER + h * D_HEAD + d] = acc1[r];
  }
}

// ---------------------------------------------------------------------------
// 8. states[n,d] = sum_t B[t,n]*exp(Alast-Acs[t])*X[t,d]   (K = 256)
// ---------------------------------------------------------------------------
__global__ __launch_bounds__(256)
void states_kernel(const bf16* __restrict__ conv, const float* __restrict__ Acs,
                   float* __restrict__ states) {
  __shared__ bf16 As[64 * 40];   // [n][t], decay-scaled
  __shared__ bf16 Xs[32 * 72];   // [t][d]
  __shared__ float dec[256];
  const int bch = blockIdx.x, ntile = blockIdx.y;
  const int h = bch & 63, bc = bch >> 6, g = h >> 3;
  const int n0 = ntile * 64;
  const float alast = Acs[(size_t)bch * CHUNK + (CHUNK - 1)];
  dec[threadIdx.x] = __expf(alast - Acs[(size_t)bch * CHUNK + threadIdx.x]);
  __syncthreads();
  const int lane = threadIdx.x & 31, wave = threadIdx.x >> 5;
  const int sn = (wave & 3) * 16, sm0 = (wave >> 2) * 32;
  v8f acc0 = V8F_ZERO, acc1 = V8F_ZERO;
  for (int k0 = 0; k0 < CHUNK; k0 += 32) {
    {
      const int r = threadIdx.x >> 3, c = (threadIdx.x & 7) * 8;
      const float ds = dec[k0 + r];
      const bf16* gb = conv + (size_t)(bc * CHUNK + k0 + r) * CONV_DIM + D_INNER + g * SN + n0 + c;
#pragma unroll
      for (int i = 0; i < 8; ++i) As[(c + i) * 40 + r] = (bf16)((float)gb[i] * ds);
    }
    {
      const int r = threadIdx.x >> 3, c = (threadIdx.x & 7) * 8;
      const bf16* gx = conv + (size_t)(bc * CHUNK + k0 + r) * CONV_DIM + h * D_HEAD + c;
#pragma unroll
      for (int i = 0; i < 8; ++i) Xs[r * 72 + c + i] = gx[i];
    }
    __syncthreads();
    v16bf bf = frag_b_kn(Xs + sn, 72, lane);
    acc0 = wmma_bf16(frag_a(As + sm0 * 40, 40, lane), bf, acc0);
    acc1 = wmma_bf16(frag_a(As + (sm0 + 16) * 40, 40, lane), bf, acc1);
    __syncthreads();
  }
  const int half = lane >> 4, nn = lane & 15, d = sn + nn;
#pragma unroll
  for (int r = 0; r < 8; ++r) {
    const int na = n0 + sm0 + half * 8 + r, nb = na + 16;
    states[(size_t)bch * (SN * D_HEAD) + (size_t)na * D_HEAD + d] = acc0[r];
    states[(size_t)bch * (SN * D_HEAD) + (size_t)nb * D_HEAD + d] = acc1[r];
  }
}

// ---------------------------------------------------------------------------
// 9. Sequential inter-chunk scan; emits h_prev (bf16) and h_final (-> d_out).
// ---------------------------------------------------------------------------
__global__ __launch_bounds__(256)
void scan_kernel(const float* __restrict__ states, const float* __restrict__ Acs,
                 bf16* __restrict__ hprev, float* __restrict__ hfinal) {
  const int bh = blockIdx.x;            // b*64 + h
  const int b = bh >> 6, h = bh & 63;
  float hreg[32];
#pragma unroll
  for (int i = 0; i < 32; ++i) hreg[i] = 0.f;
  for (int c = 0; c < NCH; ++c) {
    const int bch = (b * NCH + c) * NH + h;
    const float cd = __expf(Acs[(size_t)bch * CHUNK + (CHUNK - 1)]);
#pragma unroll
    for (int i = 0; i < 32; ++i) {
      const int idx = threadIdx.x + i * 256;
      hprev[(size_t)bch * (SN * D_HEAD) + idx] = (bf16)hreg[i];
      hreg[i] = hreg[i] * cd + states[(size_t)bch * (SN * D_HEAD) + idx];
    }
  }
#pragma unroll
  for (int i = 0; i < 32; ++i)
    hfinal[(size_t)bh * (SN * D_HEAD) + threadIdx.x + i * 256] = hreg[i];
}

// ---------------------------------------------------------------------------
// 10. Y += exp(Acs[t]) * (C[t,:] · h_prev)   (K = 128)
// ---------------------------------------------------------------------------
__global__ __launch_bounds__(256)
void ycarry_kernel(const bf16* __restrict__ conv, const bf16* __restrict__ hprev,
                   const float* __restrict__ Acs, float* __restrict__ Y) {
  __shared__ bf16 As[64 * 40];   // C tile [t][n]
  __shared__ bf16 Hs[32 * 72];   // h_prev [n][d]
  __shared__ float at[64];
  const int bch = blockIdx.x, mt = blockIdx.y;
  const int h = bch & 63, bc = bch >> 6, g = h >> 3;
  const int t0 = mt * 64;
  if (threadIdx.x < 64) at[threadIdx.x] = Acs[(size_t)bch * CHUNK + t0 + threadIdx.x];
  const int lane = threadIdx.x & 31, wave = threadIdx.x >> 5;
  const int sn = (wave & 3) * 16, sm0 = (wave >> 2) * 32;
  v8f acc0 = V8F_ZERO, acc1 = V8F_ZERO;
  for (int k0 = 0; k0 < SN; k0 += 32) {
    {
      const int r = threadIdx.x >> 2, c = (threadIdx.x & 3) * 8;
      const bf16* gc = conv + (size_t)(bc * CHUNK + t0 + r) * CONV_DIM + (D_INNER + GN) + g * SN + k0 + c;
#pragma unroll
      for (int i = 0; i < 8; ++i) As[r * 40 + c + i] = gc[i];
    }
    {
      const int r = threadIdx.x >> 3, c = (threadIdx.x & 7) * 8;
      const bf16* gh = hprev + (size_t)bch * (SN * D_HEAD) + (size_t)(k0 + r) * D_HEAD + c;
#pragma unroll
      for (int i = 0; i < 8; ++i) Hs[r * 72 + c + i] = gh[i];
    }
    __syncthreads();
    v16bf bf = frag_b_kn(Hs + sn, 72, lane);
    acc0 = wmma_bf16(frag_a(As + sm0 * 40, 40, lane), bf, acc0);
    acc1 = wmma_bf16(frag_a(As + (sm0 + 16) * 40, 40, lane), bf, acc1);
    __syncthreads();
  }
  const int half = lane >> 4, nn = lane & 15, d = sn + nn;
#pragma unroll
  for (int r = 0; r < 8; ++r) {
    const int la = sm0 + half * 8 + r, lb = la + 16;
    const size_t ia = (size_t)(bc * CHUNK + t0 + la) * D_INNER + h * D_HEAD + d;
    const size_t ib = (size_t)(bc * CHUNK + t0 + lb) * D_INNER + h * D_HEAD + d;
    Y[ia] += __expf(at[la]) * acc0[r];
    Y[ib] += __expf(at[lb]) * acc1[r];
  }
}

// ---------------------------------------------------------------------------
// 11. rmsnorm(Y, out_norm_w) * silu(gate) -> bf16
// ---------------------------------------------------------------------------
__global__ __launch_bounds__(256)
void outgate_kernel(const float* __restrict__ Y, const float* __restrict__ proj,
                    const float* __restrict__ outw, bf16* __restrict__ yn) {
  __shared__ float red[256];
  const int row = blockIdx.x;
  const float* yr = Y + (size_t)row * D_INNER;
  float ss = 0.f;
  for (int i = threadIdx.x; i < D_INNER; i += 256) { float v = yr[i]; ss += v * v; }
  red[threadIdx.x] = ss; __syncthreads();
  for (int s = 128; s > 0; s >>= 1) {
    if (threadIdx.x < s) red[threadIdx.x] += red[threadIdx.x + s];
    __syncthreads();
  }
  const float rms = rsqrtf(red[0] / (float)D_INNER + 1e-6f);
  for (int i = threadIdx.x; i < D_INNER; i += 256) {
    const float g = proj[(size_t)row * IN_PROJ + (D_INNER + NH + 2 * GN) + i];
    float v = yr[i] * rms * outw[i];
    v *= g / (1.f + __expf(-g));
    yn[(size_t)row * D_INNER + i] = (bf16)v;
  }
}

// ---------------------------------------------------------------------------
// Host launcher
// ---------------------------------------------------------------------------
extern "C" void kernel_launch(void* const* d_in, const int* in_sizes, int n_in,
                              void* d_out, int out_size, void* d_ws, size_t ws_size,
                              hipStream_t stream) {
  (void)in_sizes; (void)n_in; (void)out_size; (void)ws_size;
  const float* x          = (const float*)d_in[0];
  const float* norm_w     = (const float*)d_in[1];
  const float* in_proj_w  = (const float*)d_in[2];
  const float* conv_w     = (const float*)d_in[3];
  const float* conv_b     = (const float*)d_in[4];
  const float* conv_norm_w= (const float*)d_in[5];
  const float* A_log      = (const float*)d_in[6];
  const float* dt_w       = (const float*)d_in[7];
  const float* dt_b       = (const float*)d_in[8];
  const float* out_norm_w = (const float*)d_in[9];
  const float* out_proj_w = (const float*)d_in[10];

  float* out    = (float*)d_out;                        // [2,2048,2048]
  float* hfinal = out + (size_t)ROWS * D_MODEL;         // [2,64,128,64]

  char* p = (char*)d_ws;
  auto alloc = [&](size_t bytes) -> char* {
    char* r = p; p += (bytes + 255) & ~(size_t)255; return r;
  };
  bf16*  xn     = (bf16*) alloc((size_t)ROWS * D_MODEL * 2);
  bf16*  w_in   = (bf16*) alloc((size_t)IN_PROJ * D_MODEL * 2);
  bf16*  w_out  = (bf16*) alloc((size_t)D_MODEL * D_INNER * 2);
  float* proj   = (float*)alloc((size_t)ROWS * IN_PROJ * 4);
  bf16*  convo  = (bf16*) alloc((size_t)ROWS * CONV_DIM * 2);
  float* Aval   = (float*)alloc((size_t)ROWS * NH * 4);
  float* Acs    = (float*)alloc((size_t)BCH * CHUNK * 4);
  bf16*  Mmask  = (bf16*) alloc((size_t)BCH * CHUNK * CHUNK * 2);
  float* Ybuf   = (float*)alloc((size_t)ROWS * D_INNER * 4);
  float* states = (float*)alloc((size_t)BCH * SN * D_HEAD * 4);
  bf16*  hprev  = (bf16*) alloc((size_t)BCH * SN * D_HEAD * 2);
  bf16*  ynorm  = (bf16*) alloc((size_t)ROWS * D_INNER * 2);

  rmsnorm_x_kernel<<<ROWS, 256, 0, stream>>>(x, norm_w, xn);
  {
    size_t n = (size_t)IN_PROJ * D_MODEL;
    f32_to_bf16_kernel<<<(unsigned)((n + 255) / 256), 256, 0, stream>>>(in_proj_w, w_in, n);
  }
  {
    size_t n = (size_t)D_MODEL * D_INNER;
    f32_to_bf16_kernel<<<(unsigned)((n + 255) / 256), 256, 0, stream>>>(out_proj_w, w_out, n);
  }
  // in_proj: M=4096, N=10304, K=2048 (128x64 tiles)
  gemm_bf16_nt<<<dim3(ROWS / 128, IN_PROJ / 64), 256, 0, stream>>>(
      xn, w_in, nullptr, proj, D_MODEL, D_MODEL, D_MODEL, IN_PROJ);
  conv_norm_silu_kernel<<<ROWS, 256, 0, stream>>>(proj, conv_w, conv_b, conv_norm_w, convo);
  dt_a_kernel<<<ROWS, 64, 0, stream>>>(proj, dt_w, dt_b, A_log, Aval);
  cumsum_kernel<<<BCH, 256, 0, stream>>>(Aval, Acs);
  cb_mask_kernel<<<dim3(BCH, 16), 256, 0, stream>>>(convo, Acs, Mmask);
  ydiag_kernel<<<dim3(BCH, 4), 256, 0, stream>>>(Mmask, convo, Ybuf);
  states_kernel<<<dim3(BCH, 2), 256, 0, stream>>>(convo, Acs, states);
  scan_kernel<<<BB * NH, 256, 0, stream>>>(states, Acs, hprev, hfinal);
  ycarry_kernel<<<dim3(BCH, 4), 256, 0, stream>>>(convo, hprev, Acs, Ybuf);
  outgate_kernel<<<ROWS, 256, 0, stream>>>(Ybuf, proj, out_norm_w, ynorm);
  // out_proj: M=4096, N=2048, K=4096, fused residual add (128x64 tiles)
  gemm_bf16_nt<<<dim3(ROWS / 128, D_MODEL / 64), 256, 0, stream>>>(
      ynorm, w_out, x, out, D_INNER, D_INNER, D_INNER, D_MODEL);
}
